// GCN_26860725469234
// MI455X (gfx1250) — matrix operand order repaired and checked by reading, not compiled, not run on hardware
//
#include <hip/hip_runtime.h>
#include <cstring>
#include <cstdint>

// ---------------- problem constants ----------------
#define NN      30000        // nodes
#define EE      600000       // edges
#define EP      630000       // edges + self loops (multiple of 16)
#define CC      128          // channels
#define GG      128
#define LL      4
#define BB      128
#define CUTOFF  3.5f
#define HH1     85
#define HH2     64

typedef _Float16 f16;
typedef __attribute__((ext_vector_type(16))) _Float16 v16h;
typedef __attribute__((ext_vector_type(8)))  _Float16 v8h;
typedef __attribute__((ext_vector_type(4)))  _Float16 v4h;
typedef __attribute__((ext_vector_type(8)))  float    v8f;

__device__ __forceinline__ v8f wmma_f16(v16h a, v16h b, v8f c) {
    return __builtin_amdgcn_wmma_f32_16x16x32_f16(false, a, false, b, (short)0, c,
                                                  false, false);
}
__device__ __forceinline__ float softsign(float v) { return v / (1.f + fabsf(v)); }

// A-fragment load: two contiguous 16B chunks (k .. k+7, k+16 .. k+23) -> b128 loads
__device__ __forceinline__ v16h load_a16(const f16* p) {
    v8h lo = *(const v8h*)p;
    v8h hi = *(const v8h*)(p + 16);
    return __builtin_shufflevector(lo, hi, 0, 1, 2, 3, 4, 5, 6, 7,
                                           8, 9, 10, 11, 12, 13, 14, 15);
}

// ---------------- utility kernels ----------------
__global__ void k_cvt_f16(const float* __restrict__ a, f16* __restrict__ o, int n) {
    int t = blockIdx.x * blockDim.x + threadIdx.x;
    if (t < n) o[t] = (f16)a[t];
}
__global__ void k_fill_f32(float* p, float v, int n) {
    int t = blockIdx.x * blockDim.x + threadIdx.x;
    if (t < n) p[t] = v;
}
__global__ void k_fill_u32(unsigned* p, unsigned v, int n) {
    int t = blockIdx.x * blockDim.x + threadIdx.x;
    if (t < n) p[t] = v;
}

// pack weight W[K,N] (f32 row-major) into WMMA B-fragment-major f16 layout:
// P[((ks*ntn + nt)*32 + lane)*16 + i]  <->  K = ks*32 + (lane>=16)*8 + (i&7) + (i>>3)*16,
//                                           N = nt*16 + (lane&15)
__global__ void k_pack_b(const float* __restrict__ W, f16* __restrict__ P,
                         int Nn, int K) {
    int t = blockIdx.x * blockDim.x + threadIdx.x;
    if (t >= K * Nn) return;
    int i    = t & 15;
    int lane = (t >> 4) & 31;
    int frag = t >> 9;
    int ntn  = Nn >> 4;
    int ks   = frag / ntn, nt = frag - ks * ntn;
    int k    = ks * 32 + ((lane >> 4) << 3) + (i & 7) + ((i >> 3) << 4);
    int n    = nt * 16 + (lane & 15);
    P[t] = (f16)W[(size_t)k * Nn + n];
}

// build uniform src/dst/ew arrays including self loops
__global__ void k_edge_prep(const int* __restrict__ ei, const float* __restrict__ ea,
                            int* __restrict__ src, int* __restrict__ dst,
                            float* __restrict__ ew) {
    int e = blockIdx.x * blockDim.x + threadIdx.x;
    if (e >= EP) return;
    if (e < EE) { src[e] = ei[e]; dst[e] = ei[EE + e]; ew[e] = ea[e]; }
    else        { int n = e - EE; src[e] = n; dst[e] = n; ew[e] = 1.f; }
}

// ---------------- generic WMMA GEMM: C[M,N] = act(A[M,K]@Bpacked + bias) --------
// 4 waves/block, one wave per 16x16 tile. grid=(Nn/64, M/16), block=(32,4).
__global__ void k_gemm(const f16* __restrict__ A, const f16* __restrict__ Bp,
                       const float* __restrict__ bias,
                       float* __restrict__ Cf, f16* __restrict__ Ch,
                       int Nn, int K, int act) {
    const int lane  = threadIdx.x;
    const int tn    = blockIdx.x * blockDim.y + threadIdx.y;
    const int tm    = blockIdx.y;
    const int ntn   = Nn >> 4;
    const int cc    = lane & 15;
    const int khalf = (lane >> 4) * 8;
    const int row   = tm * 16 + cc;
    const int col   = tn * 16 + cc;
    v8f acc = {};
    for (int k0 = 0; k0 < K; k0 += 32) {
        v16h a = load_a16(A + (size_t)row * K + k0 + khalf);
        v16h b = *(const v16h*)(Bp + ((size_t)((k0 >> 5) * ntn + tn) * 32 + lane) * 16);
        acc = wmma_f16(a, b, acc);
    }
    const int orow0 = tm * 16 + (lane >> 4) * 8;
    const float bv = bias ? bias[col] : 0.f;
#pragma unroll
    for (int r = 0; r < 8; ++r) {
        float v = acc[r] + bv;
        if (act == 1) v = softsign(v);
        size_t idx = (size_t)(orow0 + r) * Nn + col;
        if (Cf) Cf[idx] = v;
        if (Ch) Ch[idx] = (f16)v;
    }
}

// ---------------- fused edge gate MLP (128 -> 64 -> 32 -> 1) -------------------
// 8 waves/block, one wave handles 16 edges; msg tile staged in LDS;
// gate weights read as packed B fragments straight from global (L0/L2 resident).
#define GW 8
__global__ void k_gate(const f16* __restrict__ m16, const int* __restrict__ src,
                       const float* __restrict__ ew,
                       const f16* __restrict__ W1p, const float* __restrict__ b1,
                       const f16* __restrict__ W2p, const float* __restrict__ b2,
                       const float* __restrict__ W3, const float* __restrict__ b3,
                       float* __restrict__ logit) {
    __shared__ f16 sMsg[GW][16 * 128];   // 32 KB
    __shared__ f16 sG1 [GW][16 * 64];    // 16 KB
    __shared__ f16 sG2 [GW][16 * 32];    //  8 KB
    const int tid = threadIdx.x;
    const int wid = tid >> 5, lane = tid & 31;
    int ebase = (blockIdx.x * GW + wid) * 16;
    if (ebase > EP - 16) ebase = EP - 16;   // clamp: duplicates recompute same values

    // stage msg = m16[src] * ew  (16 x 128 f16), vectorized 8B gather per lane/row
    for (int r = 0; r < 16; ++r) {
        int e = ebase + r;
        _Float16 w = (_Float16)ew[e];
        v4h ws = {w, w, w, w};
        v4h mv = *(const v4h*)(m16 + (size_t)src[e] * CC + lane * 4);
        *(v4h*)&sMsg[wid][r * 128 + lane * 4] = mv * ws;
    }
    __syncthreads();

    const int cc = lane & 15, khalf = (lane >> 4) * 8;
    // layer 1: 16x128 @ 128x64  (ntn=4, 4 k-steps)
#pragma unroll
    for (int nt = 0; nt < 4; ++nt) {
        v8f acc = {};
#pragma unroll
        for (int ks = 0; ks < 4; ++ks) {
            v16h a = load_a16(&sMsg[wid][cc * 128 + ks * 32 + khalf]);
            v16h b = *(const v16h*)(W1p + ((size_t)(ks * 4 + nt) * 32 + lane) * 16);
            acc = wmma_f16(a, b, acc);
        }
        int col = nt * 16 + cc;
#pragma unroll
        for (int r = 0; r < 8; ++r) {
            float v = softsign(acc[r] + b1[col]);
            sG1[wid][((lane >> 4) * 8 + r) * 64 + col] = (f16)v;
        }
    }
    __syncthreads();
    // layer 2: 16x64 @ 64x32  (ntn=2, 2 k-steps)
#pragma unroll
    for (int nt = 0; nt < 2; ++nt) {
        v8f acc = {};
#pragma unroll
        for (int ks = 0; ks < 2; ++ks) {
            v16h a = load_a16(&sG1[wid][cc * 64 + ks * 32 + khalf]);
            v16h b = *(const v16h*)(W2p + ((size_t)(ks * 2 + nt) * 32 + lane) * 16);
            acc = wmma_f16(a, b, acc);
        }
        int col = nt * 16 + cc;
#pragma unroll
        for (int r = 0; r < 8; ++r) {
            float v = softsign(acc[r] + b2[col]);
            sG2[wid][((lane >> 4) * 8 + r) * 32 + col] = (f16)v;
        }
    }
    __syncthreads();
    // layer 3: 16x32 @ 32x1  (per-row dot, lanes 0..15)
    if (lane < 16) {
        float s = b3[0];
#pragma unroll
        for (int k = 0; k < 32; ++k) s += (float)sG2[wid][lane * 32 + k] * W3[k];
        int e = ebase + lane;
        logit[e] = (ew[e] <= CUTOFF) ? s : -1e30f;
    }
}

// ---------------- segment softmax (atomics, ordered-uint max trick) -------------
__device__ __forceinline__ unsigned ord_enc(float f) {
    unsigned u = __float_as_uint(f);
    return (u >> 31) ? ~u : (u | 0x80000000u);
}
__device__ __forceinline__ float ord_dec(unsigned u) {
    return (u & 0x80000000u) ? __uint_as_float(u & 0x7FFFFFFFu) : __uint_as_float(~u);
}
__global__ void k_segmax(const float* __restrict__ logit, const int* __restrict__ dst,
                         unsigned* __restrict__ mx, int n) {
    int t = blockIdx.x * blockDim.x + threadIdx.x;
    if (t >= n) return;
    atomicMax(&mx[dst[t]], ord_enc(logit[t]));
}
__global__ void k_expsum(const float* __restrict__ logit, const int* __restrict__ dst,
                         const float* __restrict__ ew, const unsigned* __restrict__ mx,
                         float* __restrict__ earr, float* __restrict__ den, int n) {
    int t = blockIdx.x * blockDim.x + threadIdx.x;
    if (t >= n) return;
    int d = dst[t];
    float ev = (ew[t] <= CUTOFF) ? __expf(logit[t] - ord_dec(mx[d])) : 0.f;
    earr[t] = ev;
    atomicAdd(&den[d], ev);
}
// aggr[dst] += alpha * msg   (32 threads x 4 ch per edge)
__global__ void k_aggr(const float* __restrict__ m, const int* __restrict__ src,
                       const int* __restrict__ dst, const float* __restrict__ ew,
                       const float* __restrict__ earr, const float* __restrict__ den,
                       float* __restrict__ aggr, int total) {
    int t = blockIdx.x * blockDim.x + threadIdx.x;
    if (t >= total) return;
    int e = t >> 5, part = t & 31;
    int d = dst[e];
    float alpha = earr[e] / (den[d] + 1e-16f);
    if (alpha == 0.f) return;
    float w = alpha * ew[e];
    const float* ms = m + (size_t)src[e] * CC + part * 4;
    float* op = aggr + (size_t)d * CC + part * 4;
#pragma unroll
    for (int j = 0; j < 4; ++j) atomicAdd(&op[j], w * ms[j]);
}

// ---------------- GRU elementwise ----------------
__global__ void k_gru(const float* __restrict__ gi, const float* __restrict__ gh,
                      float* __restrict__ h, f16* __restrict__ h16, int n) {
    int t = blockIdx.x * blockDim.x + threadIdx.x;
    if (t >= n) return;
    int node = t >> 7, c = t & 127;
    const float* pi = gi + (size_t)node * 384;
    const float* ph = gh + (size_t)node * 384;
    float r  = 1.f / (1.f + __expf(-(pi[c] + ph[c])));
    float z  = 1.f / (1.f + __expf(-(pi[128 + c] + ph[128 + c])));
    float ng = tanhf(pi[256 + c] + r * ph[256 + c]);
    float out = (1.f - z) * ng + z * h[t];
    h[t] = out; h16[t] = (f16)out;
}

__global__ void k_cat(const f16* __restrict__ h16, const f16* __restrict__ x16,
                      f16* __restrict__ cat, int n) {
    int t = blockIdx.x * blockDim.x + threadIdx.x;
    if (t >= n) return;
    int node = t >> 8, c = t & 255;
    cat[t] = (c < 128) ? h16[(size_t)node * 128 + c] : x16[(size_t)node * 128 + (c - 128)];
}

// ---------------- attention softmax + graph pooling ----------------
__global__ void k_attn_pool(const float* __restrict__ ai2, const float* __restrict__ xj,
                            const int* __restrict__ batch, float* __restrict__ pool) {
    int n = blockIdx.x, c = threadIdx.x;
    __shared__ float red[128];
    float v = ai2[(size_t)n * GG + c];
    red[c] = v; __syncthreads();
    for (int s = 64; s > 0; s >>= 1) { if (c < s) red[c] = fmaxf(red[c], red[c + s]); __syncthreads(); }
    float mx = red[0]; __syncthreads();
    float ex = __expf(v - mx);
    red[c] = ex; __syncthreads();
    for (int s = 64; s > 0; s >>= 1) { if (c < s) red[c] += red[c + s]; __syncthreads(); }
    float attn = ex / red[0] * xj[(size_t)n * GG + c];
    atomicAdd(&pool[(size_t)batch[n] * GG + c], attn);
}

// ---------------- tiny head MLP (B=128 rows) ----------------
__global__ void k_head(const float* __restrict__ pool,
                       const float* __restrict__ W1, const float* __restrict__ b1,
                       const float* __restrict__ W2, const float* __restrict__ b2,
                       const float* __restrict__ W3, const float* __restrict__ b3,
                       float* __restrict__ out) {
    int b = threadIdx.x;
    if (b >= BB) return;
    float o1[HH1];
    for (int j = 0; j < HH1; ++j) {
        float s = b1[j];
        for (int k = 0; k < GG; ++k) s += pool[(size_t)b * GG + k] * W1[k * HH1 + j];
        o1[j] = fmaxf(s, 0.f);
    }
    float o2[HH2];
    for (int j = 0; j < HH2; ++j) {
        float s = b2[j];
        for (int k = 0; k < HH1; ++k) s += o1[k] * W2[k * HH2 + j];
        o2[j] = fmaxf(s, 0.f);
    }
    float s = b3[0];
    for (int k = 0; k < HH2; ++k) s += o2[k] * W3[k];
    out[b] = s;
}

// ---------------- host side ----------------
static inline unsigned ord_enc_host(float f) {
    unsigned u; memcpy(&u, &f, 4);
    return (u >> 31) ? ~u : (u | 0x80000000u);
}

extern "C" void kernel_launch(void* const* d_in, const int* in_sizes, int n_in,
                              void* d_out, int out_size, void* d_ws, size_t ws_size,
                              hipStream_t stream) {
    (void)in_sizes; (void)n_in; (void)out_size; (void)ws_size;
    const float* x        = (const float*)d_in[0];
    const float* edgeattr = (const float*)d_in[1];
    const float* gate_W1  = (const float*)d_in[2];
    const float* gate_b1  = (const float*)d_in[3];
    const float* gate_W2  = (const float*)d_in[4];
    const float* gate_b2  = (const float*)d_in[5];
    const float* gate_W3  = (const float*)d_in[6];
    const float* gate_b3  = (const float*)d_in[7];
    const float* ggc_W    = (const float*)d_in[8];
    const float* gru_Wih  = (const float*)d_in[9];
    const float* gru_bih  = (const float*)d_in[10];
    const float* gru_Whh  = (const float*)d_in[11];
    const float* gru_bhh  = (const float*)d_in[12];
    const float* ai_W1    = (const float*)d_in[13];
    const float* ai_b1    = (const float*)d_in[14];
    const float* ai_W2    = (const float*)d_in[15];
    const float* ai_b2    = (const float*)d_in[16];
    const float* aj_W     = (const float*)d_in[17];
    const float* aj_b     = (const float*)d_in[18];
    const float* out_W1   = (const float*)d_in[19];
    const float* out_b1   = (const float*)d_in[20];
    const float* out_W2   = (const float*)d_in[21];
    const float* out_b2   = (const float*)d_in[22];
    const float* out_W3   = (const float*)d_in[23];
    const float* out_b3   = (const float*)d_in[24];
    const int*   edge_idx = (const int*)d_in[25];
    const int*   batch    = (const int*)d_in[26];
    float* out = (float*)d_out;

    char* wp = (char*)d_ws;
    auto alloc = [&](size_t bytes) -> void* {
        void* p = (void*)wp; wp += (bytes + 255) & ~(size_t)255; return p;
    };
    int*      srcp  = (int*)alloc(4ull * EP);
    int*      dstp  = (int*)alloc(4ull * EP);
    float*    ewp   = (float*)alloc(4ull * EP);
    float*    logit = (float*)alloc(4ull * EP);
    float*    earr  = (float*)alloc(4ull * EP);
    unsigned* mx    = (unsigned*)alloc(4ull * NN);
    float*    den   = (float*)alloc(4ull * NN);
    float*    h     = (float*)alloc(4ull * NN * CC);
    float*    mbuf  = (float*)alloc(4ull * NN * CC);
    float*    aggr  = (float*)alloc(4ull * NN * CC);
    float*    ai2   = (float*)alloc(4ull * NN * GG);
    float*    xj    = (float*)alloc(4ull * NN * GG);
    float*    gi    = (float*)alloc(4ull * NN * 384);
    float*    gh    = (float*)alloc(4ull * NN * 384);
    f16*      h16   = (f16*)alloc(2ull * NN * CC);
    f16*      x16   = (f16*)alloc(2ull * NN * CC);
    f16*      m16   = (f16*)alloc(2ull * NN * CC);
    f16*      aggr16= (f16*)alloc(2ull * NN * CC);
    f16*      cat16 = (f16*)alloc(2ull * NN * 256);
    f16*      ai116 = (f16*)alloc(2ull * NN * CC);
    float*    pool  = (float*)alloc(4ull * BB * GG);
    // packed (fragment-major) f16 weights
    f16*      ggcP  = (f16*)alloc(2ull * LL * CC * CC);
    f16*      wihP  = (f16*)alloc(2ull * CC * 384);
    f16*      whhP  = (f16*)alloc(2ull * CC * 384);
    f16*      gw1P  = (f16*)alloc(2ull * 128 * 64);
    f16*      gw2P  = (f16*)alloc(2ull * 64 * 32);
    f16*      aiw1P = (f16*)alloc(2ull * 256 * 128);
    f16*      aiw2P = (f16*)alloc(2ull * 128 * 128);
    f16*      ajwP  = (f16*)alloc(2ull * 128 * 128);

    const int T = 256;
    auto cd = [](int a, int b) { return (a + b - 1) / b; };

    // pack weights into WMMA fragment layout (f32 -> f16 + shuffle, once)
    for (int l = 0; l < LL; ++l)
        k_pack_b<<<cd(CC*CC,T),T,0,stream>>>(ggc_W + (size_t)l*CC*CC,
                                             ggcP + (size_t)l*CC*CC, CC, CC);
    k_pack_b<<<cd(CC*384,T),T,0,stream>>>(gru_Wih, wihP, 384, CC);
    k_pack_b<<<cd(CC*384,T),T,0,stream>>>(gru_Whh, whhP, 384, CC);
    k_pack_b<<<cd(128*64,T),T,0,stream>>>(gate_W1, gw1P, 64, 128);
    k_pack_b<<<cd(64*32,T),T,0,stream>>>(gate_W2,  gw2P, 32, 64);
    k_pack_b<<<cd(256*128,T),T,0,stream>>>(ai_W1,  aiw1P, 128, 256);
    k_pack_b<<<cd(128*128,T),T,0,stream>>>(ai_W2,  aiw2P, 128, 128);
    k_pack_b<<<cd(128*128,T),T,0,stream>>>(aj_W,   ajwP, 128, 128);
    k_cvt_f16<<<cd(NN*CC,T),T,0,stream>>>(x, x16, NN*CC);

    // h = x ; h16 = x16
    hipMemcpyAsync(h,   x,   4ull*NN*CC, hipMemcpyDeviceToDevice, stream);
    hipMemcpyAsync(h16, x16, 2ull*NN*CC, hipMemcpyDeviceToDevice, stream);

    k_edge_prep<<<cd(EP,T),T,0,stream>>>(edge_idx, edgeattr, srcp, dstp, ewp);

    const unsigned NEG_INF_ENC = ord_enc_host(-3.0e38f);
    const dim3 blk(32, 4);
    const dim3 gN128(GG/64, NN/16);     // (2,1875) : 4 tn-tiles per block
    const dim3 gN384(384/64, NN/16);    // (6,1875)

    for (int l = 0; l < LL; ++l) {
        // m = h @ ggc_W[l]            (WMMA, f32 + f16 copies)
        k_gemm<<<gN128,blk,0,stream>>>(h16, ggcP + (size_t)l*CC*CC, nullptr,
                                       mbuf, m16, CC, CC, 0);
        // fused edge gate MLP -> logit
        k_gate<<<cd(EP/16, GW), GW*32, 0, stream>>>(m16, srcp, ewp,
                                                    gw1P, gate_b1, gw2P, gate_b2,
                                                    gate_W3, gate_b3, logit);
        // segment softmax + aggregate
        k_fill_u32<<<cd(NN,T),T,0,stream>>>(mx, NEG_INF_ENC, NN);
        k_fill_f32<<<cd(NN,T),T,0,stream>>>(den, 0.f, NN);
        k_fill_f32<<<cd(NN*CC,T),T,0,stream>>>(aggr, 0.f, NN*CC);
        k_segmax<<<cd(EP,T),T,0,stream>>>(logit, dstp, mx, EP);
        k_expsum<<<cd(EP,T),T,0,stream>>>(logit, dstp, ewp, mx, earr, den, EP);
        k_aggr<<<cd(EP*32,T),T,0,stream>>>(mbuf, srcp, dstp, ewp, earr, den, aggr, EP*32);
        k_cvt_f16<<<cd(NN*CC,T),T,0,stream>>>(aggr, aggr16, NN*CC);
        // GRU gates (two WMMA GEMMs) + elementwise update
        k_gemm<<<gN384,blk,0,stream>>>(aggr16, wihP, gru_bih, gi, nullptr, 384, CC, 0);
        k_gemm<<<gN384,blk,0,stream>>>(h16,    whhP, gru_bhh, gh, nullptr, 384, CC, 0);
        k_gru<<<cd(NN*CC,T),T,0,stream>>>(gi, gh, h, h16, NN*CC);
    }

    // readout
    k_cat<<<cd(NN*256,T),T,0,stream>>>(h16, x16, cat16, NN*256);
    k_gemm<<<gN128,blk,0,stream>>>(cat16, aiw1P, ai_b1, nullptr, ai116, CC, 256, 1);
    k_gemm<<<gN128,blk,0,stream>>>(ai116, aiw2P, ai_b2, ai2,     nullptr, CC, CC, 1);
    k_gemm<<<gN128,blk,0,stream>>>(x16,   ajwP,  aj_b,  xj,      nullptr, CC, CC, 1);
    k_fill_f32<<<cd(BB*GG,T),T,0,stream>>>(pool, 0.f, BB*GG);
    k_attn_pool<<<NN,128,0,stream>>>(ai2, xj, batch, pool);
    k_head<<<1,128,0,stream>>>(pool, out_W1, out_b1, out_W2, out_b2, out_W3, out_b3, out);
}
